// GridModel_F_25658134626580
// MI455X (gfx1250) — matrix operand order repaired and checked by reading, not compiled, not run on hardware
//
#include <hip/hip_runtime.h>
#include <hip/hip_bf16.h>
#include <math.h>

// ---------------------------------------------------------------------------
// Grid-cell (MEC/HPC) attractor step for MI455X (gfx1250).
// Bandwidth-bound: conn_in (328MB) + conn_mat (400MB) streams dominate.
// Irec GEMM routed through v_wmma_f32_16x16x32_bf16 (M padded 8->16).
// ---------------------------------------------------------------------------

#define PI_F      3.14159265358979323846f
#define TWO_PI_F  6.28318530717958647692f

typedef __attribute__((ext_vector_type(16))) __bf16 v16bf;
typedef __attribute__((ext_vector_type(8)))  float  v8f;
typedef __attribute__((ext_vector_type(4)))  float  fvec4;   // native vector for NT loads

constexpr int MMOD = 8;
constexpr int NM   = 100;
constexpr int NUM  = 10000;      // neurons per module
constexpr int NHPC = 1024;
constexpr int LDK  = 10016;      // K padded to multiple of 32 for WMMA

// ws layout (float offsets)
constexpr int INP_OFF  = 0;         // 80000 : input_pos + hpc feedback
constexpr int IREC_OFF = 80000;     // 80000 : recurrent input accumulator
constexpr int SUM_OFF  = 160000;    // 1     : global sum of r1
constexpr int A_OFF_F  = 160008;    // bf16 A matrix [16][LDK] (uint16)

// d_out layout (floats): r_new 80000 | u1 80000 | v1 80000 | center 16 | ci 16 | pe 16
constexpr int OUT_U1 = 80000, OUT_V1 = 160000;
constexpr int OUT_C = 240000, OUT_CI = 240016, OUT_PE = 240032;

__device__ __forceinline__ float pbound(float d) {   // floored-mod period bound
    float t = fmodf(d + PI_F, TWO_PI_F);
    t = (t < 0.0f) ? t + TWO_PI_F : t;
    return t - PI_F;
}

__device__ __forceinline__ unsigned short f2bf(float f) {  // f32 -> bf16 RNE
    unsigned u = __float_as_uint(f);
    u += 0x7FFFu + ((u >> 16) & 1u);
    return (unsigned short)(u >> 16);
}

#if __has_builtin(__builtin_amdgcn_cvt_pk_bf16_f32)
__device__ __forceinline__ unsigned int packbf(float f0, float f1) {
    auto t = __builtin_amdgcn_cvt_pk_bf16_f32(f0, f1);   // v_cvt_pk_bf16_f32 (RNE)
    unsigned int r;
    __builtin_memcpy(&r, &t, 4);
    return r;
}
#else
__device__ __forceinline__ unsigned int packbf(float f0, float f1) {
    unsigned u0 = __float_as_uint(f0); u0 += 0x7FFFu + ((u0 >> 16) & 1u);
    unsigned u1 = __float_as_uint(f1); u1 += 0x7FFFu + ((u1 >> 16) & 1u);
    return (u0 >> 16) | ((u1 >> 16) << 16);
}
#endif

// ---------------------------------------------------------------------------
// Kernel A: per-module small math + input_pos + bf16 A-matrix staging.
// One block per module (8 blocks x 256 threads).
// ---------------------------------------------------------------------------
__global__ __launch_bounds__(256) void kA(const float* __restrict__ pos,
                                          const float* __restrict__ vel,
                                          const float* __restrict__ r,
                                          const float* __restrict__ pe,
                                          const float* __restrict__ ci,
                                          float* __restrict__ out,
                                          float* __restrict__ ws) {
    const int m   = blockIdx.x;
    const int tid = threadIdx.x;
    const float* rm = r + m * NUM;

    // module constants
    const float ratio = 1.2f + (float)m * (10.0f - 1.2f) / 7.0f;
    const float p2p   = ratio * TWO_PI_F / 10.0f;
    const float ang   = (float)m * (PI_F / 3.0f) / 7.0f;
    const float cs = cosf(ang), sn = sinf(ang);
    const float inv_s3 = 0.57735026918962576451f;       // 1/sqrt(3)
    const float s3h    = 0.86602540378443864676f;       // sqrt(3)/2

    // Phase from pos (fmod — truncated remainder, matches jnp.fmod)
    const float lx  = (cs * pos[0] - sn * pos[1]) * p2p;
    const float ly  = (sn * pos[0] + cs * pos[1]) * p2p;
    const float phx = fmodf(lx - ly * inv_s3 + PI_F, TWO_PI_F) - PI_F;
    const float phy = fmodf(ly * (2.0f * inv_s3) + PI_F, TWO_PI_F) - PI_F;

    const float gw = PI_F * 0.5f;
    const float inv_gw2 = 1.0f / (gw * gw);
    const float step = TWO_PI_F / (float)NM;

    unsigned short* A16 = (unsigned short*)(ws + A_OFF_F);
    float* inp = ws + INP_OFF;

    float lmax = -1e30f;
    for (int n = tid; n < NUM; n += 256) {
        const float xg = -PI_F + (float)(n % NM) * step;
        const float yg = -PI_F + (float)(n / NM) * step;
        const float dxp = pbound(phx - xg);
        const float dyp = pbound(phy - yg);
        const float t0 = dxp + 0.5f * dyp;
        const float t1 = s3h * dyp;
        const float d2 = t0 * t0 + t1 * t1;
        inp[m * NUM + n] = 10.0f * __expf(-0.25f * d2 * inv_gw2);
        const float rv = rm[n];
        lmax = fmaxf(lmax, rv);
        A16[m * LDK + n]       = f2bf(rv);   // real row
        A16[(m + 8) * LDK + n] = 0;          // M-pad row
    }
    for (int n = NUM + tid; n < LDK; n += 256) {   // K-pad tail
        A16[m * LDK + n]       = 0;
        A16[(m + 8) * LDK + n] = 0;
    }

    __shared__ float sd[256];
    sd[tid] = lmax; __syncthreads();
    for (int s = 128; s > 0; s >>= 1) {
        if (tid < s) sd[tid] = fmaxf(sd[tid], sd[tid + s]);
        __syncthreads();
    }
    const float thr = 0.1f * sd[0];
    __syncthreads();

    float acc[4] = {0.f, 0.f, 0.f, 0.f};   // Ssin_x, Scos_x, Ssin_y, Scos_y
    for (int n = tid; n < NUM; n += 256) {
        const float rv = rm[n];
        if (rv > thr) {
            const float xg = -PI_F + (float)(n % NM) * step;
            const float yg = -PI_F + (float)(n / NM) * step;
            acc[0] += rv * __sinf(xg); acc[1] += rv * __cosf(xg);
            acc[2] += rv * __sinf(yg); acc[3] += rv * __cosf(yg);
        }
    }
    float tot[4];
    for (int q = 0; q < 4; q++) {
        sd[tid] = acc[q]; __syncthreads();
        for (int s = 128; s > 0; s >>= 1) {
            if (tid < s) sd[tid] += sd[tid + s];
            __syncthreads();
        }
        tot[q] = sd[0]; __syncthreads();
    }

    if (tid == 0) {
        const float cx = atan2f(tot[0], tot[1]);
        const float cy = atan2f(tot[2], tot[3]);
        out[OUT_C + 2 * m]     = cx;
        out[OUT_C + 2 * m + 1] = cy;
        // v_phase
        float vrx = (cs * vel[0] - sn * vel[1]) * p2p;
        float vry = (sn * vel[0] + cs * vel[1]) * p2p;
        const float vpx = vrx - vry * inv_s3;
        const float vpy = vry * (2.0f * inv_s3);
        out[OUT_CI + 2 * m]     = pbound(ci[2 * m]     + vpx);
        out[OUT_CI + 2 * m + 1] = pbound(ci[2 * m + 1] + vpy);
        const float dpx = pbound(cx - pe[2 * m]);
        const float dpy = pbound(cy - pe[2 * m + 1]);
        out[OUT_PE + 2 * m]     = pbound(pe[2 * m]     + dpx * 0.001f + vpx);
        out[OUT_PE + 2 * m + 1] = pbound(pe[2 * m + 1] + dpy * 0.001f + vpy);
    }
}

// ---------------------------------------------------------------------------
// Kernel B: conn_in GEMV (328MB NT stream), one row per wave32.
// Fuses relu(.-0.1)*0.5 into inp; zeros Irec and the r1-sum accumulator.
// ---------------------------------------------------------------------------
__global__ __launch_bounds__(256) void kB(const float* __restrict__ conn_in,
                                          const float* __restrict__ r_hpc,
                                          float* __restrict__ ws) {
    if (blockIdx.x == 0 && threadIdx.x == 0) ws[SUM_OFF] = 0.0f;
    const int wid  = (blockIdx.x * 256 + threadIdx.x) >> 5;   // row in [0,80000)
    const int lane = threadIdx.x & 31;

    const fvec4* ci4 = (const fvec4*)(conn_in + (size_t)wid * NHPC);
    const fvec4* rh4 = (const fvec4*)r_hpc;
    float acc = 0.0f;
#pragma unroll
    for (int k = 0; k < 8; k++) {
        const fvec4 a = __builtin_nontemporal_load(&ci4[lane + 32 * k]); // read-once
        const fvec4 b = rh4[lane + 32 * k];                              // hot, cached
        acc += a.x * b.x + a.y * b.y + a.z * b.z + a.w * b.w;
    }
#pragma unroll
    for (int off = 16; off > 0; off >>= 1) acc += __shfl_xor(acc, off);
    if (lane == 0) {
        ws[INP_OFF + wid] += fmaxf(acc - 0.1f, 0.0f) * 0.5f;   // hpc2mec_stre
        ws[IREC_OFF + wid] = 0.0f;
    }
}

// ---------------------------------------------------------------------------
// Kernel C: Irec = A(16 x 10016 bf16) x conn_mat(10016 x 10000, f32->bf16 OTF)
// 625 column strips x 16 K-chunks = 10000 waves; v_wmma_f32_16x16x32_bf16.
// Partial 16x16 tiles merged with global_atomic_add_f32 (rows 0-7 only).
// ---------------------------------------------------------------------------
template<bool CLAMP>
__device__ __forceinline__ void kc_step(const unsigned int* __restrict__ ap,
                                        const float* __restrict__ bp,
                                        const float* __restrict__ conn_mat,
                                        int j, int kb, v8f& c) {
    union { unsigned int u[8]; v16bf v; } a, b;
    // A operand: lane li holds row M=li; K = 16*h + s (contiguous, matches B)
#pragma unroll
    for (int p = 0; p < 4; p++) { a.u[p] = ap[p]; a.u[4 + p] = ap[4 + p]; }
    // B operand: 32x16 tile of conn_mat, column j, bf16-packed on the fly
#pragma unroll
    for (int p = 0; p < 8; p++) {
        float f0, f1;
        if (CLAMP) {                       // K-pad tail only (A is zero there)
            int K0 = kb + 2 * p;
            int K1 = K0 + 1;
            K0 = (K0 < NUM) ? K0 : NUM - 1;
            K1 = (K1 < NUM) ? K1 : NUM - 1;
            f0 = __builtin_nontemporal_load(&conn_mat[K0 * NUM + j]);
            f1 = __builtin_nontemporal_load(&conn_mat[K1 * NUM + j]);
        } else {                           // constant offsets fit 24-bit IOFFSET
            f0 = __builtin_nontemporal_load(bp + (2 * p) * NUM);
            f1 = __builtin_nontemporal_load(bp + (2 * p + 1) * NUM);
        }
        b.u[p] = packbf(f0, f1);
    }
    c = __builtin_amdgcn_wmma_f32_16x16x32_bf16(false, a.v, false, b.v,
                                                (short)0, c, false, false);
}

__global__ __launch_bounds__(256) void kC(const float* __restrict__ conn_mat,
                                          float* __restrict__ ws) {
    const int lane  = threadIdx.x & 31;
    const int w     = (blockIdx.x * 256 + threadIdx.x) >> 5;  // 0..9999 exactly
    const int strip = w % 625;
    const int kc    = w / 625;                                // 0..15
    const int li    = lane & 15;
    const int h     = lane >> 4;
    const int j     = strip * 16 + li;                        // output column
    const int k0    = kc * 640;
    const int k1    = (k0 + 640 < LDK) ? (k0 + 640) : LDK;
    const int ksafe = (k1 <= NUM) ? k1 : (NUM & ~31);         // last full in-bounds tile

    const unsigned int* A32 = (const unsigned int*)(ws + A_OFF_F);
    const unsigned int* ap = A32 + ((li * LDK + k0 + 16 * h) >> 1);
    const float*        bp = conn_mat + (size_t)(k0 + 16 * h) * NUM + j;

    v8f c = {};
    int k = k0;
#pragma unroll 2
    for (; k < ksafe; k += 32) {                              // hot loop: no clamps
        kc_step<false>(ap, bp, conn_mat, j, 0, c);
        __builtin_prefetch(bp + 32 * NUM, 0, 1);              // next tile -> caches
        ap += 16;
        bp += 32 * NUM;
    }
    for (; k < k1; k += 32) {                                 // <=1 clamped tail iter
        kc_step<true>(ap, bp, conn_mat, j, k + 16 * h, c);
        ap += 16;
        bp += 32 * NUM;
    }

    if (h == 0) {   // lanes 0-15 hold D rows M=0..7 (real modules)
        float* Irec = ws + IREC_OFF;
#pragma unroll
        for (int p = 0; p < 8; p++) atomicAdd(&Irec[p * NUM + j], c[p]);
    }
}

// ---------------------------------------------------------------------------
// Kernel D: fused Euler step + r1 = u1^2 + global sum of r1.
// ---------------------------------------------------------------------------
__global__ __launch_bounds__(256) void kD(const float* __restrict__ u,
                                          const float* __restrict__ v,
                                          const float* __restrict__ st,
                                          float* __restrict__ out,
                                          float* __restrict__ ws) {
    const int idx = blockIdx.x * 256 + threadIdx.x;
    const float dt = st[1] - st[0];
    float r1 = 0.0f;
    if (idx < MMOD * NUM) {
        const float it = ws[INP_OFF + idx] + ws[IREC_OFF + idx];
        const float uu = u[idx], vv = v[idx];
        const float u1 = fmaxf(uu + dt * (-uu + it - vv), 0.0f);   // tau = 1
        const float v1 = vv + dt * (-vv + 0.1f * uu) * 0.1f;       // tau_v = 10
        out[OUT_U1 + idx] = u1;
        out[OUT_V1 + idx] = v1;
        r1 = u1 * u1;
    }
    __shared__ float sd[256];
    sd[threadIdx.x] = r1; __syncthreads();
    for (int s = 128; s > 0; s >>= 1) {
        if (threadIdx.x < s) sd[threadIdx.x] += sd[threadIdx.x + s];
        __syncthreads();
    }
    if (threadIdx.x == 0) atomicAdd(ws + SUM_OFF, sd[0]);
}

// ---------------------------------------------------------------------------
// Kernel E: divisive normalization.
// ---------------------------------------------------------------------------
__global__ __launch_bounds__(256) void kE(float* __restrict__ out,
                                          const float* __restrict__ ws) {
    const int idx = blockIdx.x * 256 + threadIdx.x;
    if (idx < MMOD * NUM) {
        const float u1 = out[OUT_U1 + idx];
        const float S  = ws[SUM_OFF];
        out[idx] = (u1 * u1) / (1.0f + 0.002f * S);    // k = 20/NUM
    }
}

// ---------------------------------------------------------------------------
extern "C" void kernel_launch(void* const* d_in, const int* in_sizes, int n_in,
                              void* d_out, int out_size, void* d_ws, size_t ws_size,
                              hipStream_t stream) {
    const float* pos      = (const float*)d_in[0];
    const float* velocity = (const float*)d_in[1];
    const float* r_hpc    = (const float*)d_in[2];
    const float* shared_t = (const float*)d_in[3];
    const float* r        = (const float*)d_in[4];
    const float* u        = (const float*)d_in[5];
    const float* v        = (const float*)d_in[6];
    const float* pe       = (const float*)d_in[7];
    const float* ci       = (const float*)d_in[8];
    const float* conn_in  = (const float*)d_in[9];
    const float* conn_mat = (const float*)d_in[10];
    float* out = (float*)d_out;
    float* ws  = (float*)d_ws;

    kA<<<MMOD, 256, 0, stream>>>(pos, velocity, r, pe, ci, out, ws);
    kB<<<(MMOD * NUM) / 8, 256, 0, stream>>>(conn_in, r_hpc, ws);   // 10000 blocks
    kC<<<1250, 256, 0, stream>>>(conn_mat, ws);                      // 10000 waves
    kD<<<(MMOD * NUM + 255) / 256, 256, 0, stream>>>(u, v, shared_t, out, ws);
    kE<<<(MMOD * NUM + 255) / 256, 256, 0, stream>>>(out, ws);
}
